// Attention_78108275245493
// MI455X (gfx1250) — matrix operand order repaired and checked by reading, not compiled, not run on hardware
//
#include <hip/hip_runtime.h>
#include <stdint.h>

// ---------------------------------------------------------------------------
// Cross-attention (B=2,N=2048,DIM=1024,H=16,DH=64, J=2048+1+256=2305) for
// MI455X / gfx1250.  All GEMMs + both attention matmuls run on
// v_wmma_f32_16x16x32_bf16 (bf16 in, fp32 accumulate) -- 8x the per-issue K
// of the f32 WMMA path.  Softmax / LayerNorm statistics stay fp32.
// Round-2 change: __launch_bounds__ to stop Q-fragment spills in the
// attention hot loop (disasm showed scratch_load_b128 reloads per K-chunk).
// ---------------------------------------------------------------------------

typedef __attribute__((ext_vector_type(16))) __bf16 v16bf;
typedef __attribute__((ext_vector_type(8)))  float  v8f;

#define DEVINL __device__ __forceinline__

static constexpr int BB   = 2;
static constexpr int NN   = 2048;
static constexpr int DIM  = 1024;
static constexpr int HH   = 16;
static constexpr int DH   = 64;
static constexpr int CTXN = 256;
static constexpr int JV   = NN + 1 + CTXN;   // 2305 valid keys
static constexpr int JP   = 2336;            // padded to multiple of 32

DEVINL uint16_t f2bf(float x) {
  uint32_t u = __builtin_bit_cast(uint32_t, x);
  uint32_t r = (u + 0x7FFFu + ((u >> 16) & 1u)) >> 16;   // round-nearest-even
  return (uint16_t)r;
}

union FragU { uint32_t u[8]; v16bf v; };

// A-fragment (16x32 bf16, ISA 7.12.2): row = lane%16,
// k = (i<4?0:16) + 8*(lane/16) + 2*(i&3).  Source row-major, lda elements.
DEVINL v16bf load_a(const uint16_t* __restrict__ p, int lda) {
  int lane = threadIdx.x & 31;
  int row  = lane & 15;
  int lh   = lane >> 4;
  FragU f;
#pragma unroll
  for (int i = 0; i < 8; ++i) {
    int k = ((i < 4) ? 0 : 16) + lh * 8 + 2 * (i & 3);
    f.u[i] = *(const uint32_t*)(p + (size_t)row * lda + k);
  }
  return f.v;
}

// B-fragment (32x16 bf16): n = lane%16, k = (i<4?0:16) + 8*(lane/16) + 2*(i&3).
// Element (k,n) lives at p[n*ldn + k]  (source is N-major with K contiguous
// -> packed dword loads).
DEVINL v16bf load_b(const uint16_t* __restrict__ p, int ldn) {
  int lane = threadIdx.x & 31;
  int n    = lane & 15;
  int lh   = lane >> 4;
  FragU f;
#pragma unroll
  for (int i = 0; i < 8; ++i) {
    int k = ((i < 4) ? 0 : 16) + lh * 8 + 2 * (i & 3);
    f.u[i] = *(const uint32_t*)(p + (size_t)n * ldn + k);
  }
  return f.v;
}

// ---------------------------------------------------------------------------
// fp32 [K][N] -> bf16 [N][K] (weights, once; makes B-fragments dword-
// contiguous).
__global__ __launch_bounds__(256) void cvt_transpose_kernel(
    const float* __restrict__ W, uint16_t* __restrict__ Wt, int K, int N) {
  int t = blockIdx.x * blockDim.x + threadIdx.x;
  if (t >= K * N) return;
  int k = t % K, n = t / K;
  Wt[t] = f2bf(W[(size_t)k * N + n]);
}

// ---------------------------------------------------------------------------
// LayerNorm over 1024 cols (population variance), optional bias, bf16 or f32
// output.
template <bool OUT_BF16>
__global__ __launch_bounds__(256) void ln_kernel(
    const float* __restrict__ in, const float* __restrict__ gamma,
    const float* __restrict__ beta, void* __restrict__ out, float eps) {
  const int C = 1024;
  int row = blockIdx.x, tid = threadIdx.x;
  const float* rp = in + (size_t)row * C;
  float v[4], s1 = 0.f, s2 = 0.f;
#pragma unroll
  for (int i = 0; i < 4; ++i) { v[i] = rp[tid + i * 256]; s1 += v[i]; s2 += v[i] * v[i]; }
  __shared__ float r1[256], r2[256];
  r1[tid] = s1; r2[tid] = s2;
  __syncthreads();
  for (int st = 128; st > 0; st >>= 1) {
    if (tid < st) { r1[tid] += r1[tid + st]; r2[tid] += r2[tid + st]; }
    __syncthreads();
  }
  float mean = r1[0] * (1.f / C);
  float var  = r2[0] * (1.f / C) - mean * mean;
  float rstd = rsqrtf(var + eps);
#pragma unroll
  for (int i = 0; i < 4; ++i) {
    int c = tid + i * 256;
    float y = (v[i] - mean) * rstd * gamma[c] + (beta ? beta[c] : 0.f);
    if constexpr (OUT_BF16) ((uint16_t*)out)[(size_t)row * C + c] = f2bf(y);
    else                    ((float*)out)[(size_t)row * C + c] = y;
  }
}

// ---------------------------------------------------------------------------
// C[M,Nc] (+bias, *alpha) = A[M,K]bf16 @ Wt[Nc,K]bf16 ; 8 waves/block,
// 32x64 block tile, one 16x16 WMMA tile per wave.
template <bool OUT_BF16>
__global__ __launch_bounds__(256, 1) void gemm_bf16_kernel(
    const uint16_t* __restrict__ A, const uint16_t* __restrict__ Wt,
    const float* __restrict__ bias, void* __restrict__ Cout,
    int Nc, int K, float alpha) {
  int wave = threadIdx.x >> 5;
  int rowbase = blockIdx.x * 32 + (wave >> 2) * 16;
  int colbase = blockIdx.y * 64 + (wave & 3) * 16;
  int lane = threadIdx.x & 31;
  v8f acc = {};
  const uint16_t* ap = A  + (size_t)rowbase * K;
  const uint16_t* bp = Wt + (size_t)colbase * K;
  for (int k0 = 0; k0 < K; k0 += 32) {
    v16bf a = load_a(ap + k0, K);
    v16bf b = load_b(bp + k0, K);
    acc = __builtin_amdgcn_wmma_f32_16x16x32_bf16(false, a, false, b,
                                                  (short)0, acc, false, false);
  }
  int lh  = lane >> 4;
  int col = colbase + (lane & 15);
  float bv = bias ? bias[col] : 0.f;
#pragma unroll
  for (int r = 0; r < 8; ++r) {
    int row = rowbase + r + 8 * lh;
    float y = acc[r] * alpha + bv;
    if constexpr (OUT_BF16) ((uint16_t*)Cout)[(size_t)row * Nc + col] = f2bf(y);
    else                    ((float*)Cout)[(size_t)row * Nc + col] = y;
  }
}

// ---------------------------------------------------------------------------
// Assemble K [b][JP][64] and V^T [b][64][JP] from the x-derived KV rows, the
// null token, and the context KV rows; zero the padded tail.
__global__ __launch_bounds__(256) void pack_kv_kernel(
    const uint16_t* __restrict__ kvtmp, const uint16_t* __restrict__ ckvtmp,
    const float* __restrict__ null_kv, uint16_t* __restrict__ Kmat,
    uint16_t* __restrict__ Vt) {
  int t = blockIdx.x * blockDim.x + threadIdx.x;
  if (t >= BB * JP * DH) return;
  int d = t & 63;
  int j = (t >> 6) % JP;
  int b = t / (JP * DH);
  uint16_t kv = 0, vv = 0;
  if (j < NN) {
    size_t o = ((size_t)(b * NN + j)) * 128;
    kv = kvtmp[o + d]; vv = kvtmp[o + 64 + d];
  } else if (j == NN) {
    kv = f2bf(null_kv[d]); vv = f2bf(null_kv[64 + d]);
  } else if (j < JV) {
    size_t o = ((size_t)(b * CTXN + (j - NN - 1))) * 128;
    kv = ckvtmp[o + d]; vv = ckvtmp[o + 64 + d];
  }
  Kmat[((size_t)(b * JP + j)) * 64 + d] = kv;
  Vt[(size_t)b * 64 * JP + (size_t)d * JP + j] = vv;
}

// ---------------------------------------------------------------------------
// Flash attention: one wave owns 16 queries x 1 head, streams all JP keys in
// chunks of 32.  P tile is staged through LDS to convert C-layout -> A-layout
// (same-wave DS RAW covered by s_wait_dscnt 0; DS ops are in-order per wave
// so the loop-carried WAR is safe).  __launch_bounds__(128,1): let the RA
// keep Q fragments + O accumulators resident (~110 VGPRs live).
__global__ __launch_bounds__(128, 1) void attn_kernel(
    const uint16_t* __restrict__ Q, const uint16_t* __restrict__ Kmat,
    const uint16_t* __restrict__ Vt, uint16_t* __restrict__ AO) {
  int w    = threadIdx.x >> 5;
  int wid  = blockIdx.x * 4 + w;
  int qt   = wid & 127;          // query tile (N/16 = 128)
  int hh   = (wid >> 7) & 15;    // head
  int b    = wid >> 11;          // batch
  int lane = threadIdx.x & 31;
  int lh   = lane >> 4;
  int cn   = lane & 15;

  __shared__ uint16_t pTile[4][16 * 32];
  uint16_t* pt = pTile[w];

  int rowbase = b * NN + qt * 16;
  const uint16_t* qp = Q + (size_t)rowbase * DIM + hh * DH;
  v16bf qa0 = load_a(qp, DIM);        // d = 0..31  (scale already folded in)
  v16bf qa1 = load_a(qp + 32, DIM);   // d = 32..63

  const uint16_t* Kb = Kmat + (size_t)b * JP * 64;
  const uint16_t* Vb = Vt   + (size_t)b * 64 * JP;

  float m[8], l[8];
#pragma unroll
  for (int r = 0; r < 8; ++r) { m[r] = -3.0e38f; l[r] = 0.f; }
  v8f o0 = {}, o1 = {}, o2 = {}, o3 = {};

  for (int jc = 0; jc < JP; jc += 32) {
    if (jc + 64 < JP) __builtin_prefetch(Kb + (size_t)(jc + 64) * 64, 0, 1);

    // sim tiles: s0 -> keys jc..jc+15, s1 -> keys jc+16..jc+31
    v8f s0 = {}, s1 = {};
    {
      v16bf b0 = load_b(Kb + (size_t)jc * 64, 64);
      v16bf b1 = load_b(Kb + (size_t)jc * 64 + 32, 64);
      s0 = __builtin_amdgcn_wmma_f32_16x16x32_bf16(false, qa0, false, b0, (short)0, s0, false, false);
      s0 = __builtin_amdgcn_wmma_f32_16x16x32_bf16(false, qa1, false, b1, (short)0, s0, false, false);
    }
    {
      v16bf b0 = load_b(Kb + (size_t)(jc + 16) * 64, 64);
      v16bf b1 = load_b(Kb + (size_t)(jc + 16) * 64 + 32, 64);
      s1 = __builtin_amdgcn_wmma_f32_16x16x32_bf16(false, qa0, false, b0, (short)0, s1, false, false);
      s1 = __builtin_amdgcn_wmma_f32_16x16x32_bf16(false, qa1, false, b1, (short)0, s1, false, false);
    }

    // mask padded keys (per-lane column index, uniform across regs)
    if (jc + cn >= JV) {
#pragma unroll
      for (int r = 0; r < 8; ++r) s0[r] = -1.0e30f;
    }
    if (jc + 16 + cn >= JV) {
#pragma unroll
      for (int r = 0; r < 8; ++r) s1[r] = -1.0e30f;
    }

    // online softmax (row = r + 8*lh; reduce across the 16 lanes of a half),
    // write P to LDS inside the same loop to keep live ranges short.
    float al[8];
#pragma unroll
    for (int r = 0; r < 8; ++r) {
      float rm = fmaxf(s0[r], s1[r]);
      rm = fmaxf(rm, __shfl_xor(rm, 1));
      rm = fmaxf(rm, __shfl_xor(rm, 2));
      rm = fmaxf(rm, __shfl_xor(rm, 4));
      rm = fmaxf(rm, __shfl_xor(rm, 8));
      float mn = fmaxf(m[r], rm);
      al[r] = __expf(m[r] - mn);
      m[r]  = mn;
      float p0 = __expf(s0[r] - mn);
      float p1 = __expf(s1[r] - mn);
      int row = r + 8 * lh;
      pt[row * 32 + cn]      = f2bf(p0);
      pt[row * 32 + 16 + cn] = f2bf(p1);
      float rs = p0 + p1;
      rs += __shfl_xor(rs, 1);
      rs += __shfl_xor(rs, 2);
      rs += __shfl_xor(rs, 4);
      rs += __shfl_xor(rs, 8);
      l[r] = l[r] * al[r] + rs;
    }
#pragma unroll
    for (int r = 0; r < 8; ++r) { o0[r] *= al[r]; o1[r] *= al[r]; o2[r] *= al[r]; o3[r] *= al[r]; }

    asm volatile("s_wait_dscnt 0" ::: "memory");
    v16bf pa;
    {
      FragU f;
#pragma unroll
      for (int i = 0; i < 8; ++i) {
        int k = ((i < 4) ? 0 : 16) + lh * 8 + 2 * (i & 3);
        f.u[i] = *(const uint32_t*)(pt + cn * 32 + k);
      }
      pa = f.v;
    }

    // O += P @ V   (V^T layout: element (k=j, n=d) at Vb[n*JP + jc + k])
    o0 = __builtin_amdgcn_wmma_f32_16x16x32_bf16(false, pa, false, load_b(Vb + (size_t)0 * 16 * JP + jc, JP), (short)0, o0, false, false);
    o1 = __builtin_amdgcn_wmma_f32_16x16x32_bf16(false, pa, false, load_b(Vb + (size_t)1 * 16 * JP + jc, JP), (short)0, o1, false, false);
    o2 = __builtin_amdgcn_wmma_f32_16x16x32_bf16(false, pa, false, load_b(Vb + (size_t)2 * 16 * JP + jc, JP), (short)0, o2, false, false);
    o3 = __builtin_amdgcn_wmma_f32_16x16x32_bf16(false, pa, false, load_b(Vb + (size_t)3 * 16 * JP + jc, JP), (short)0, o3, false, false);
  }

  // normalize and write [b,n,h*64+d] as bf16 for the output projection
#pragma unroll
  for (int r = 0; r < 8; ++r) {
    float inv = 1.f / l[r];
    size_t orow = (size_t)(rowbase + r + 8 * lh) * DIM + hh * DH;
    AO[orow + 0 * 16 + cn] = f2bf(o0[r] * inv);
    AO[orow + 1 * 16 + cn] = f2bf(o1[r] * inv);
    AO[orow + 2 * 16 + cn] = f2bf(o2[r] * inv);
    AO[orow + 3 * 16 + cn] = f2bf(o3[r] * inv);
  }
}

// ---------------------------------------------------------------------------
extern "C" void kernel_launch(void* const* d_in, const int* in_sizes, int n_in,
                              void* d_out, int out_size, void* d_ws, size_t ws_size,
                              hipStream_t stream) {
  (void)in_sizes; (void)n_in; (void)out_size; (void)ws_size;
  const float* x     = (const float*)d_in[0];
  const float* ctx   = (const float*)d_in[1];
  const float* g1    = (const float*)d_in[2];
  const float* Wq    = (const float*)d_in[3];
  const float* Wkv   = (const float*)d_in[4];
  const float* nkv   = (const float*)d_in[5];
  const float* ctx_g = (const float*)d_in[6];
  const float* ctx_b = (const float*)d_in[7];
  const float* Wc    = (const float*)d_in[8];
  const float* bc    = (const float*)d_in[9];
  const float* Wout  = (const float*)d_in[10];
  const float* g2    = (const float*)d_in[11];

  char* ws = (char*)d_ws;
  size_t off = 0;
  auto alloc = [&](size_t bytes) -> void* {
    void* p = ws + off;
    off = (off + bytes + 255) & ~(size_t)255;
    return p;
  };
  uint16_t* WqT    = (uint16_t*)alloc((size_t)1024 * 1024 * 2);
  uint16_t* WkvT   = (uint16_t*)alloc((size_t)128  * 1024 * 2);
  uint16_t* WcT    = (uint16_t*)alloc((size_t)128  * 1024 * 2);
  uint16_t* WoutT  = (uint16_t*)alloc((size_t)1024 * 1024 * 2);
  uint16_t* hbuf   = (uint16_t*)alloc((size_t)BB * NN * DIM * 2);
  uint16_t* chat   = (uint16_t*)alloc((size_t)BB * CTXN * DIM * 2);
  uint16_t* qbuf   = (uint16_t*)alloc((size_t)BB * NN * DIM * 2);
  uint16_t* kvtmp  = (uint16_t*)alloc((size_t)BB * NN * 128 * 2);
  uint16_t* ckvtmp = (uint16_t*)alloc((size_t)BB * CTXN * 128 * 2);
  uint16_t* Kmat   = (uint16_t*)alloc((size_t)BB * JP * 64 * 2);
  uint16_t* Vtb    = (uint16_t*)alloc((size_t)BB * 64 * JP * 2);
  uint16_t* attnout= (uint16_t*)alloc((size_t)BB * NN * DIM * 2);
  float*    yout   = (float*)   alloc((size_t)BB * NN * DIM * 4);

  // 1) weights -> bf16, transposed to [N][K]
  cvt_transpose_kernel<<<(1024 * 1024 + 255) / 256, 256, 0, stream>>>(Wq,   WqT,   1024, 1024);
  cvt_transpose_kernel<<<(1024 * 128  + 255) / 256, 256, 0, stream>>>(Wkv,  WkvT,  1024, 128);
  cvt_transpose_kernel<<<(1024 * 128  + 255) / 256, 256, 0, stream>>>(Wc,   WcT,   1024, 128);
  cvt_transpose_kernel<<<(1024 * 1024 + 255) / 256, 256, 0, stream>>>(Wout, WoutT, 1024, 1024);

  // 2) LayerNorms -> bf16 activations
  ln_kernel<true><<<BB * NN,   256, 0, stream>>>(x,   g1,    nullptr, hbuf, 1e-5f);
  ln_kernel<true><<<BB * CTXN, 256, 0, stream>>>(ctx, ctx_g, ctx_b,   chat, 1e-6f);

  // 3) projections (scale 1/sqrt(64) folded into Q)
  gemm_bf16_kernel<true><<<dim3(128, 16), 256, 0, stream>>>(hbuf, WqT,  nullptr, qbuf,   1024, 1024, 0.125f);
  gemm_bf16_kernel<true><<<dim3(128, 2),  256, 0, stream>>>(hbuf, WkvT, nullptr, kvtmp,  128,  1024, 1.0f);
  gemm_bf16_kernel<true><<<dim3(16, 2),   256, 0, stream>>>(chat, WcT,  bc,      ckvtmp, 128,  1024, 1.0f);

  // 4) splice null + context KV tokens, pad, build V^T
  pack_kv_kernel<<<(BB * JP * DH + 255) / 256, 256, 0, stream>>>(kvtmp, ckvtmp, nkv, Kmat, Vtb);

  // 5) flash attention: 2*16*128 = 4096 waves, 4 per block
  attn_kernel<<<1024, 128, 0, stream>>>(qbuf, Kmat, Vtb, attnout);

  // 6) output projection (fp32 out) + final LayerNorm -> d_out
  gemm_bf16_kernel<false><<<dim3(128, 16), 256, 0, stream>>>(attnout, WoutT, nullptr, yout, 1024, 1024, 1.0f);
  ln_kernel<false><<<BB * NN, 256, 0, stream>>>(yout, g2, nullptr, d_out, 1e-5f);
}
